// Self_Atten_22273700397143
// MI455X (gfx1250) — compile-verified, matrix operand order checked by reading, not run on hardware
//
#include <hip/hip_runtime.h>

#define B_   4
#define C_   512
#define CK_  64
#define N_   4096
#define MT_  (2*CK_ + C_)   // 640 projection output rows (f:64, g:64, h:512)

typedef __attribute__((ext_vector_type(8)))  float    v8f;
typedef __attribute__((ext_vector_type(8)))  _Float16 v8h;
typedef __attribute__((ext_vector_type(16))) _Float16 v16h;

// Build a 16x16x32 f16 WMMA operand for one lane.
// ISA 16-bit A/B layout: lane holds K = base+ko..ko+7 (VGPR0..3) and
// K = base+ko+16..ko+23 (VGPR4..7), ko = (lane>>4)*8. Two aligned 16B loads.
static __device__ __forceinline__ v16h ld_ab(const _Float16* p) {
  v8h lo = *(const v8h*)(p);
  v8h hi = *(const v8h*)(p + 16);
  return __builtin_shufflevector(lo, hi, 0, 1, 2, 3, 4, 5, 6, 7,
                                         8, 9, 10, 11, 12, 13, 14, 15);
}

static __device__ __forceinline__ v8f wmma16(v16h a, v16h b, v8f c) {
  // (neg_a, A, neg_b, B, c_mod, C, reuse_a, reuse_b)
  return __builtin_amdgcn_wmma_f32_16x16x32_f16(false, a, false, b,
                                                (short)0, c, false, false);
}

// Async global -> LDS copy, 16B per lane (tracked by ASYNCcnt).
// VDST = LDS byte address (low 32 bits of the generic shared pointer),
// VADDR = 64-bit global address, saddr = off (GV mode).
static __device__ __forceinline__ void async_g2l_b128(unsigned lds_off,
                                                      const _Float16* gaddr) {
  asm volatile("global_load_async_to_lds_b128 %0, %1, off"
               :: "v"(lds_off), "v"(gaddr) : "memory");
}
static __device__ __forceinline__ void wait_async0() {
  asm volatile("s_wait_asynccnt 0x0" ::: "memory");
}

// ---------------------------------------------------------------------------
// Kernel 0a: concat + convert weights [640 x 512] fp32 -> f16
// rows 0..63 = Wf, 64..127 = Wg, 128..639 = Wh
// ---------------------------------------------------------------------------
__global__ __launch_bounds__(256) void cvt_weights(
    const float* __restrict__ Wf, const float* __restrict__ Wg,
    const float* __restrict__ Wh, _Float16* __restrict__ Wc) {
  int idx = blockIdx.x * 256 + threadIdx.x;
  if (idx >= MT_ * C_) return;
  int m = idx >> 9;          // /512
  int c = idx & (C_ - 1);
  float v;
  if (m < CK_)            v = Wf[m * C_ + c];
  else if (m < 2 * CK_)   v = Wg[(m - CK_) * C_ + c];
  else                    v = Wh[(m - 2 * CK_) * C_ + c];
  Wc[idx] = (_Float16)v;
}

// ---------------------------------------------------------------------------
// Kernel 0b: transpose+convert x[b, c, n] fp32 -> xT[b, n, c] f16
// 64x64 tiles via LDS; both global phases coalesced.
// ---------------------------------------------------------------------------
__global__ __launch_bounds__(256) void xpose_x(
    const float* __restrict__ x, _Float16* __restrict__ xT) {
  __shared__ _Float16 tsh[64][72];
  const int tid = threadIdx.x;
  const int n0 = blockIdx.x * 64, c0 = blockIdx.y * 64, b = blockIdx.z;
  const float* xb = x + (size_t)b * C_ * N_;
  for (int e = tid; e < 64 * 64; e += 256) {
    int cr = e >> 6, nl = e & 63;                 // coalesced along n
    tsh[cr][nl] = (_Float16)xb[(size_t)(c0 + cr) * N_ + n0 + nl];
  }
  __syncthreads();
  _Float16* yb = xT + (size_t)b * N_ * C_;
  for (int e = tid; e < 64 * 64; e += 256) {
    int nl = e >> 6, cl = e & 63;                 // coalesced along c
    yb[(size_t)(n0 + nl) * C_ + c0 + cl] = tsh[cl][nl];
  }
}

// ---------------------------------------------------------------------------
// Kernel 1: projections, no LDS — every operand is two contiguous 16B loads.
//   rows m in [0,128): f/g -> computed as D(M=n, N=m) = xT * Wc^T and stored
//                      TRANSPOSED: fT/gT[b][n][k] (k contiguous, coalesced)
//   rows m in [128,640): h -> D(M=m, N=n) = Wc * x, stored h[b][c][n]
// Grid (N/256, MT/16, B), 512 threads = 16 waves, wave w owns n-subtile w.
// ---------------------------------------------------------------------------
__global__ __launch_bounds__(512) void proj_kernel(
    const _Float16* __restrict__ xT, const _Float16* __restrict__ Wc,
    const float* __restrict__ biasf, const float* __restrict__ biasg,
    const float* __restrict__ biash,
    _Float16* __restrict__ fT, _Float16* __restrict__ gT,
    _Float16* __restrict__ h) {
  const int tid = threadIdx.x, lane = tid & 31, w = tid >> 5;
  const int col = lane & 15;
  const int ko = (lane >> 4) * 8;
  const int mhalf = (lane >> 4) * 8;
  const int b = blockIdx.z;
  const int n0 = blockIdx.x * 256 + w * 16;
  const int mbase = blockIdx.y * 16;

  const _Float16* wrow = Wc + (size_t)(mbase + col) * C_;            // row m
  const _Float16* xrow = xT + ((size_t)b * N_ + n0 + col) * C_;      // row n

  v8f acc = {};
  if (mbase < 2 * CK_) {
    // f/g: swap operands so D rows are n -> transposed store is coalesced
    #pragma unroll
    for (int k0 = 0; k0 < C_; k0 += 32) {
      v16h a  = ld_ab(xrow + k0 + ko);   // A: M=n, K=c contiguous
      v16h bb = ld_ab(wrow + k0 + ko);   // B: N=m, K=c contiguous
      acc = wmma16(a, bb, acc);
    }
    const int   isF  = (mbase < CK_);
    const float bv   = isF ? biasf[mbase + col] : biasg[mbase - CK_ + col];
    const int   kcol = (isF ? mbase : mbase - CK_) + col;
    _Float16* yb = (isF ? fT : gT) + (size_t)b * N_ * CK_;
    #pragma unroll
    for (int r = 0; r < 8; ++r) {
      int n = n0 + r + mhalf;
      yb[(size_t)n * CK_ + kcol] = (_Float16)(acc[r] + bv);
    }
  } else {
    #pragma unroll
    for (int k0 = 0; k0 < C_; k0 += 32) {
      v16h a  = ld_ab(wrow + k0 + ko);   // A: M=m(c-out), K=c contiguous
      v16h bb = ld_ab(xrow + k0 + ko);   // B: N=n, K=c contiguous
      acc = wmma16(a, bb, acc);
    }
    _Float16* yb = h + (size_t)b * C_ * N_;
    #pragma unroll
    for (int r = 0; r < 8; ++r) {
      int m = mbase - 2 * CK_ + r + mhalf;
      yb[(size_t)m * N_ + n0 + col] = (_Float16)(acc[r] + biash[m]);
    }
  }
}

// ---------------------------------------------------------------------------
// Kernel 2: fused flash attention + residual.
// Grid (N/64, B), 512 threads (16 waves). Row block I = 64, col block J = 64.
//   g tile: double-buffered in LDS, filled by global_load_async_to_lds_b128
//           (issued after the S-store barrier, overlapped with softmax,
//            awaited with s_wait_asynccnt before barrier 5, ahead of the O GEMM).
//   S = f_I^T g_J : A hoisted to registers (J-invariant), B from LDS.
//   online softmax: 512 threads, 8 per row, two-stage LDS reduction.
//   O^T[c,i] += h[c,j] P[i,j] : A = h 16B global loads, B = Psh (f16) in LDS.
//   out = gamma * (O/l) + x, stores coalesced along i.
// ---------------------------------------------------------------------------
__global__ __launch_bounds__(512) void attn_kernel(
    const _Float16* __restrict__ fT, const _Float16* __restrict__ gT,
    const _Float16* __restrict__ h, const float* __restrict__ x,
    const float* __restrict__ gamma, float* __restrict__ out) {
  __shared__ _Float16 gsh[2][64][72]; // double-buffered g tile [j][k]
  __shared__ float    Ssh[64][66];    // raw logits [i][j]
  __shared__ _Float16 Psh[64][72];    // f16 probabilities [i][j], 144B rows
  __shared__ float    red[64][8];     // per-row partial max/sum
  __shared__ float    row_mx[64], m_s[64], l_s[64], sc_s[64];

  const int tid = threadIdx.x, lane = tid & 31, w = tid >> 5;
  const int b = blockIdx.y, i0 = blockIdx.x * 64;
  const int it = w & 3, jt = w >> 2;
  const int col = lane & 15;
  const int ko = (lane >> 4) * 8;
  const int mhalf = (lane >> 4) * 8;
  const int cbase = w * 32;

  const _Float16* fb = fT + (size_t)b * N_ * CK_;
  const _Float16* gb = gT + (size_t)b * N_ * CK_;
  const _Float16* hb = h + (size_t)b * C_ * N_;

  // hoist f_I operand for this wave's S tile (J-invariant)
  const _Float16* frow = fb + (size_t)(i0 + it * 16 + col) * CK_;
  v16h aS0 = ld_ab(frow + 0 + ko);
  v16h aS1 = ld_ab(frow + 32 + ko);

  if (tid < 64) { m_s[tid] = -3.0e38f; l_s[tid] = 0.0f; }

  v8f zero = {};
  v8f accO[2][4];
  #pragma unroll
  for (int ct = 0; ct < 2; ++ct)
    #pragma unroll
    for (int iv = 0; iv < 4; ++iv) accO[ct][iv] = zero;

  // prologue: async-stage first g tile (8KB, 16B per lane), then wait
  const int arow = tid >> 3, achk = (tid & 7) * 8;
  async_g2l_b128((unsigned)(size_t)&gsh[0][arow][achk],
                 gb + (size_t)arow * CK_ + achk);
  wait_async0();
  __syncthreads();

  for (int j0 = 0; j0 < N_; j0 += 64) {
    const int buf = (j0 >> 6) & 1;

    if (j0 + 64 < N_ && tid < 64)            // global_prefetch_b8 next h cols
      __builtin_prefetch(hb + (size_t)(tid * 8) * N_ + j0 + 64, 0, 1);

    // S tile: A = f^T (M=i, K=k) [registers], B = g (K=k, N=j) [LDS]
    const _Float16* gl = &gsh[buf][jt * 16 + col][0];
    v8f s = zero;
    s = wmma16(aS0, ld_ab(gl + 0 + ko), s);
    s = wmma16(aS1, ld_ab(gl + 32 + ko), s);
    #pragma unroll
    for (int r = 0; r < 8; ++r)
      Ssh[it * 16 + r + mhalf][jt * 16 + col] = s[r];
    __syncthreads();                         // barrier 1: S done, buf^1 idle

    // kick off async fill of the other buffer for the next J block
    if (j0 + 64 < N_)
      async_g2l_b128((unsigned)(size_t)&gsh[buf ^ 1][arow][achk],
                     gb + (size_t)(j0 + 64 + arow) * CK_ + achk);

    // online softmax, 8 threads per row
    {
      int r = tid >> 3, seg = tid & 7;
      float mx = -3.0e38f;
      #pragma unroll
      for (int j = 0; j < 8; ++j) mx = fmaxf(mx, Ssh[r][seg * 8 + j]);
      red[r][seg] = mx;
    }
    __syncthreads();                         // barrier 2
    if (tid < 64) {
      float mx = m_s[tid];
      #pragma unroll
      for (int s2 = 0; s2 < 8; ++s2) mx = fmaxf(mx, red[tid][s2]);
      sc_s[tid] = __expf(m_s[tid] - mx);     // 0 on first block
      row_mx[tid] = mx;
      m_s[tid] = mx;
    }
    __syncthreads();                         // barrier 3
    {
      int r = tid >> 3, seg = tid & 7;
      float mx = row_mx[r];
      float sum = 0.0f;
      #pragma unroll
      for (int j = 0; j < 8; ++j) {
        float p = __expf(Ssh[r][seg * 8 + j] - mx);
        Psh[r][seg * 8 + j] = (_Float16)p;
        sum += p;
      }
      red[r][seg] = sum;
    }
    __syncthreads();                         // barrier 4
    if (tid < 64) {
      float sum = 0.0f;
      #pragma unroll
      for (int s2 = 0; s2 < 8; ++s2) sum += red[tid][s2];
      l_s[tid] = l_s[tid] * sc_s[tid] + sum;
    }
    wait_async0();                           // own async writes complete
    __syncthreads();                         // barrier 5: all asyncs visible

    // rescale accumulators (N-dim of O tiles is i -> scale depends on col)
    #pragma unroll
    for (int iv = 0; iv < 4; ++iv) {
      float sc = sc_s[iv * 16 + col];
      #pragma unroll
      for (int r = 0; r < 8; ++r) { accO[0][iv][r] *= sc; accO[1][iv][r] *= sc; }
    }

    // O^T += h * P^T : A = h (M=c, K=j) 16B global loads, B = P (K=j, N=i)
    #pragma unroll
    for (int k0 = 0; k0 < 64; k0 += 32) {
      v16h a0 = ld_ab(hb + (size_t)(cbase +      col) * N_ + j0 + k0 + ko);
      v16h a1 = ld_ab(hb + (size_t)(cbase + 16 + col) * N_ + j0 + k0 + ko);
      #pragma unroll
      for (int iv = 0; iv < 4; ++iv) {
        v16h bb = ld_ab(&Psh[iv * 16 + col][k0 + ko]);
        accO[0][iv] = wmma16(a0, bb, accO[0][iv]);
        accO[1][iv] = wmma16(a1, bb, accO[1][iv]);
      }
    }
    __syncthreads();
  }

  // epilogue: out[b,c,i] = gamma * O[c,i]/l[i] + x[b,c,i]; coalesced along i
  const float gamma0 = gamma[0];
  const float* xb = x + (size_t)b * C_ * N_;
  float*       ob = out + (size_t)b * C_ * N_;
  #pragma unroll
  for (int iv = 0; iv < 4; ++iv) {
    float inv = 1.0f / l_s[iv * 16 + col];
    #pragma unroll
    for (int ct = 0; ct < 2; ++ct) {
      #pragma unroll
      for (int r = 0; r < 8; ++r) {
        int c = cbase + ct * 16 + r + mhalf;
        int i = i0 + iv * 16 + col;
        size_t idx = (size_t)c * N_ + i;
        ob[idx] = gamma0 * (accO[ct][iv][r] * inv) + xb[idx];
      }
    }
  }
}

// ---------------------------------------------------------------------------
extern "C" void kernel_launch(void* const* d_in, const int* in_sizes, int n_in,
                              void* d_out, int out_size, void* d_ws, size_t ws_size,
                              hipStream_t stream) {
  (void)in_sizes; (void)n_in; (void)out_size; (void)ws_size;
  const float* x   = (const float*)d_in[0];
  const float* Wf  = (const float*)d_in[1];
  const float* bfp = (const float*)d_in[2];
  const float* Wg  = (const float*)d_in[3];
  const float* bgp = (const float*)d_in[4];
  const float* Wh  = (const float*)d_in[5];
  const float* bhp = (const float*)d_in[6];
  const float* gam = (const float*)d_in[7];
  float*       outp = (float*)d_out;

  // ws (f16): Wc 640KB | xT 16MB | fT 2MB | gT 2MB | h 16MB  (~37MB)
  _Float16* Wc = (_Float16*)d_ws;
  _Float16* xT = Wc + (size_t)MT_ * C_;
  _Float16* fT = xT + (size_t)B_ * N_ * C_;
  _Float16* gT = fT + (size_t)B_ * N_ * CK_;
  _Float16* h  = gT + (size_t)B_ * N_ * CK_;

  cvt_weights<<<(MT_ * C_ + 255) / 256, 256, 0, stream>>>(Wf, Wg, Wh, Wc);
  xpose_x<<<dim3(N_ / 64, C_ / 64, B_), 256, 0, stream>>>(x, xT);
  proj_kernel<<<dim3(N_ / 256, MT_ / 16, B_), 512, 0, stream>>>(
      xT, Wc, bfp, bgp, bhp, fT, gT, h);
  attn_kernel<<<dim3(N_ / 64, B_), 512, 0, stream>>>(fT, gT, h, x, gam, outp);
}
